// Model_34789235097649
// MI455X (gfx1250) — matrix operand order repaired and checked by reading, not compiled, not run on hardware
//
#include <hip/hip_runtime.h>
#include <math.h>

typedef __attribute__((ext_vector_type(16))) _Float16 v16h;
typedef __attribute__((ext_vector_type(8)))  float    v8f;
typedef __attribute__((ext_vector_type(2)))  float    v2f;

#define BATCH      256
#define NPRIM      32
#define PATOMS     8
#define NCLS       10
#define DATOMS     16
#define IN_DIM     1152      // 32*36
#define M1         102400    // B*20*20
#define K1         96        // 81 padded
#define K2         20736     // 256*81
#define M2         9216      // B*36

// --- CDNA5 async Global->LDS copy (ASYNCcnt-tracked DMA, bypasses VGPRs) ----
#if __has_builtin(__builtin_amdgcn_global_load_async_to_lds_b128) && \
    __has_builtin(__builtin_amdgcn_s_wait_asynccnt)
#define USE_ASYNC 1
typedef int int4v __attribute__((vector_size(16)));
typedef __attribute__((address_space(1))) int4v* gas_i4p;
typedef __attribute__((address_space(3))) int4v* las_i4p;
#define ASYNC_CP16(dst, src) \
    __builtin_amdgcn_global_load_async_to_lds_b128((gas_i4p)(src), (las_i4p)(dst), 0, 0)
#define ASYNC_WAIT() __builtin_amdgcn_s_wait_asynccnt(0)
#else
#define USE_ASYNC 0
#define ASYNC_CP16(dst, src) (*(float4*)(dst) = *(const float4*)(src))
#define ASYNC_WAIT() ((void)0)
#endif

// ---------------------------------------------------------------------------
// Generic f16 WMMA GEMM:  C[M,N] = A[M,K] * Bt[N,K]^T   (Bt stored row-major [N][K])
// AMODE 0: A is row-major f16 [M][K]
// AMODE 1: implicit im2col from NHWC f16 h[B][20][20][256], K = (ky*9+kx)*256+c
// EPI 0: raw f32 store    EPI 1: bias+relu -> f16    EPI 3: bias+sigmoid -> f32 (guard n<Nstore)
// ---------------------------------------------------------------------------
template<int AMODE, int EPI>
__global__ __launch_bounds__(256)
void gemm_f16(const _Float16* __restrict__ A, const _Float16* __restrict__ Bt,
              const float* __restrict__ bias, float* __restrict__ Cf,
              _Float16* __restrict__ Ch, int M, int N, int K, int ldc, int Nstore)
{
    __shared__ _Float16 As[2][128][40];
    __shared__ _Float16 Bs[2][64][40];

    const int tid  = threadIdx.x;
    const int m0   = blockIdx.x * 128;
    const int n0   = blockIdx.y * 64;
    const int wave = tid >> 5;
    const int lane = tid & 31;
    const int wm   = wave & 3;   // 4 waves along M (32 rows each)
    const int wn   = wave >> 2;  // 2 waves along N (32 cols each)
    const int nk   = K >> 5;

    v8f c00 = {}, c01 = {}, c10 = {}, c11 = {};

    auto loadTiles = [&](int kt, int sbuf) {
        const int k0 = kt << 5;
        // A tile: 128 rows x 32 halves, 4 threads/row x 8 halves, 2 passes
        #pragma unroll
        for (int l = 0; l < 2; ++l) {
            const int r = (tid >> 2) + (l << 6);
            const int q = tid & 3;
            const _Float16* src;
            if (AMODE == 0) {
                src = A + (size_t)(m0 + r) * K + k0 + q * 8;
            } else {
                const int m  = m0 + r;
                const int b  = m / 36, s = m % 36;
                const int oy = s / 6,  ox = s % 6;
                const int kk = k0 + q * 8;
                const int pos = kk >> 8, c = kk & 255;
                const int ky = pos / 9, kx = pos % 9;
                src = A + ((((size_t)b * 20 + (2 * oy + ky)) * 20 + (2 * ox + kx)) * 256 + c);
            }
            ASYNC_CP16(&As[sbuf][r][q * 8], src);
        }
        // B tile (transposed source): 64 n-rows x 32 halves
        {
            const int nr = tid >> 2;
            const int q  = tid & 3;
            const _Float16* src = Bt + (size_t)(n0 + nr) * K + k0 + q * 8;
            __builtin_prefetch(src + 32, 0, 1);   // speculative prefetch of next K-chunk
            ASYNC_CP16(&Bs[sbuf][nr][q * 8], src);
        }
    };

    loadTiles(0, 0);
    int s = 0;
    for (int kt = 0; kt < nk; ++kt) {
        ASYNC_WAIT();        // my wave's async tile DMA complete (no-op in sync path)
        __syncthreads();     // all waves' pieces visible in LDS
        if (kt + 1 < nk) loadTiles(kt + 1, s ^ 1);   // overlap next-tile DMA with WMMAs

        // fragments per documented wave32 VGPR layouts
        const int rA  = wm * 32 + (lane & 15);
        const int kb  = (lane < 16) ? 0 : 8;
        const int nB  = wn * 32 + (lane & 15);
        const int kb2 = (lane < 16) ? 0 : 16;

        union Frag { v16h v; float4 f[2]; } a0, a1, b0, b1;
        a0.f[0] = *(const float4*)&As[s][rA][kb];
        a0.f[1] = *(const float4*)&As[s][rA][kb + 16];
        a1.f[0] = *(const float4*)&As[s][rA + 16][kb];
        a1.f[1] = *(const float4*)&As[s][rA + 16][kb + 16];
        b0.f[0] = *(const float4*)&Bs[s][nB][kb];
        b0.f[1] = *(const float4*)&Bs[s][nB][kb + 16];
        b1.f[0] = *(const float4*)&Bs[s][nB + 16][kb];
        b1.f[1] = *(const float4*)&Bs[s][nB + 16][kb + 16];

        c00 = __builtin_amdgcn_wmma_f32_16x16x32_f16(false, a0.v, false, b0.v, (short)0, c00, false, false);
        c01 = __builtin_amdgcn_wmma_f32_16x16x32_f16(false, a0.v, false, b1.v, (short)0, c01, false, false);
        c10 = __builtin_amdgcn_wmma_f32_16x16x32_f16(false, a1.v, false, b0.v, (short)0, c10, false, false);
        c11 = __builtin_amdgcn_wmma_f32_16x16x32_f16(false, a1.v, false, b1.v, (short)0, c11, false, false);
        s ^= 1;
    }

    const int nloc = wn * 32 + (lane & 15);
    const int mloc = wm * 32 + ((lane < 16) ? 0 : 8);
    auto store = [&](v8f& c, int mt, int nt) {
        const int n = n0 + nloc + nt * 16;
        #pragma unroll
        for (int j = 0; j < 8; ++j) {
            const int m = m0 + mloc + mt * 16 + j;
            const float v = c[j];
            if (EPI == 0) {
                Cf[(size_t)m * ldc + n] = v;
            } else if (EPI == 1) {
                const float t = v + bias[n];
                Ch[(size_t)m * ldc + n] = (_Float16)(t > 0.f ? t : 0.f);
            } else {
                if (n < Nstore) {
                    const float t = v + bias[n];
                    Cf[(size_t)m * ldc + n] = 1.f / (1.f + expf(-t));
                }
            }
        }
    };
    store(c00, 0, 0); store(c01, 0, 1); store(c10, 1, 0); store(c11, 1, 1);
}

// ---------------------------------------------------------------------------
// Digit-cap votes: per i (1152): votes[b,(o,t)] = x3[b,i,:8] @ digit_w[i,:8,(o,t)]
// fp32 WMMA 16x16x4, one wave per (i, mtile, ntile) -> 1152*16*10 waves
// ---------------------------------------------------------------------------
__global__ __launch_bounds__(256)
void votes_wmma(const float* __restrict__ x3, const float* __restrict__ dw,
                float* __restrict__ votes)
{
    const int tid  = threadIdx.x;
    const int wave = tid >> 5;
    const int lane = tid & 31;
    const long job = (long)blockIdx.x * 8 + wave;
    const int nt = (int)(job % 10);
    const int mt = (int)((job / 10) & 15);
    const int i  = (int)(job / 160);

    const int r   = lane & 15;
    const int col = nt * 16 + r;
    v8f c = {};
    #pragma unroll
    for (int k4 = 0; k4 < 2; ++k4) {
        const int ka = k4 * 4 + ((lane < 16) ? 0 : 2);
        v2f a, b;
        const int bm = mt * 16 + r;
        const float* ap = x3 + ((size_t)bm * IN_DIM + i) * 8 + ka;
        a[0] = ap[0]; a[1] = ap[1];
        const float* bp = dw + ((size_t)i * 8 + ka) * 160 + col;
        b[0] = bp[0]; b[1] = bp[160];
        c = __builtin_amdgcn_wmma_f32_16x16x4_f32(false, a, false, b, (short)0, c, false, false);
    }
    const int mb = mt * 16 + ((lane < 16) ? 0 : 8);
    #pragma unroll
    for (int j = 0; j < 8; ++j)
        votes[((size_t)(mb + j) * IN_DIM + i) * 160 + col] = c[j];
}

// ---------------------------------------------------------------------------
// Routing iteration: one workgroup per batch element; route softmax staged in LDS
// ---------------------------------------------------------------------------
__global__ __launch_bounds__(512)
void routing_iter(const float* __restrict__ votes, const float* __restrict__ digit_b,
                  float* __restrict__ logits, float* __restrict__ act_out,
                  int update_logits)
{
    __shared__ float route[IN_DIM * NCLS];
    __shared__ float pre[160];
    __shared__ float scale_s[NCLS];

    const int b   = blockIdx.x;
    const int tid = threadIdx.x;
    const float* vb = votes  + (size_t)b * IN_DIM * 160;
    float*       lb = logits + (size_t)b * IN_DIM * NCLS;

    for (int i = tid; i < IN_DIM; i += 512) {
        float l[NCLS], mx = -1e30f;
        #pragma unroll
        for (int o = 0; o < NCLS; ++o) { l[o] = lb[i * NCLS + o]; mx = fmaxf(mx, l[o]); }
        float sum = 0.f;
        #pragma unroll
        for (int o = 0; o < NCLS; ++o) { l[o] = expf(l[o] - mx); sum += l[o]; }
        const float inv = 1.f / sum;
        #pragma unroll
        for (int o = 0; o < NCLS; ++o) route[i * NCLS + o] = l[o] * inv;
    }
    __syncthreads();

    if (tid < 160) {
        const int o = tid >> 4;
        float acc = digit_b[tid];
        for (int i = 0; i < IN_DIM; ++i)
            acc += route[i * NCLS + o] * vb[(size_t)i * 160 + tid];
        pre[tid] = acc;
    }
    __syncthreads();
    if (tid < NCLS) {
        float n2 = 0.f;
        #pragma unroll
        for (int t = 0; t < 16; ++t) { const float u = pre[tid * 16 + t]; n2 += u * u; }
        scale_s[tid] = (n2 / (1.f + n2)) * rsqrtf(n2 + 1e-9f);
    }
    __syncthreads();
    if (tid < 160) {
        const float a = pre[tid] * scale_s[tid >> 4];
        pre[tid] = a;
        act_out[(size_t)b * 160 + tid] = a;
    }
    __syncthreads();
    if (update_logits) {
        for (int p = tid; p < IN_DIM * NCLS; p += 512) {
            const int i = p / NCLS, o = p % NCLS;
            const float* vv = vb + (size_t)i * 160 + o * 16;
            float d = 0.f;
            #pragma unroll
            for (int t = 0; t < 16; ++t) d += vv[t] * pre[o * 16 + t];
            lb[p] += d;
        }
    }
}

// ---------------------------------------------------------------------------
// Elementwise helper kernels
// ---------------------------------------------------------------------------
__global__ void k_zero_f32(float* p, int n) {
    int i = blockIdx.x * 256 + threadIdx.x;
    if (i < n) p[i] = 0.f;
}
// im2col for conv1: A1[m][k], m=(b,y,x) over 20x20, k = ky*9+kx (pad 81->96)
__global__ void k_im2col1(const float* __restrict__ x, _Float16* __restrict__ A1) {
    long idx = (long)blockIdx.x * 256 + threadIdx.x;
    if (idx >= (long)M1 * K1) return;
    const int k = (int)(idx % K1);
    const int m = (int)(idx / K1);
    const int b = m / 400, s = m % 400;
    const int y = s / 20, xx = s % 20;
    float v = 0.f;
    if (k < 81) v = x[(size_t)b * 784 + (y + k / 9) * 28 + (xx + k % 9)];
    A1[idx] = (_Float16)v;
}
// conv1 weights -> Bt [256][96]
__global__ void k_cvt_w1(const float* __restrict__ w, _Float16* __restrict__ Bt) {
    int idx = blockIdx.x * 256 + threadIdx.x;
    if (idx >= 256 * K1) return;
    const int o = idx / K1, k = idx % K1;
    Bt[idx] = (_Float16)((k < 81) ? w[o * 81 + k] : 0.f);
}
// primary weights OIHW -> Bt [256][20736], k = (ky*9+kx)*256 + c
__global__ void k_cvt_wp(const float* __restrict__ w, _Float16* __restrict__ Bt) {
    long idx = (long)blockIdx.x * 256 + threadIdx.x;
    if (idx >= (long)256 * K2) return;
    const int o = (int)(idx / K2), k = (int)(idx % K2);
    const int p = k >> 8, c = k & 255;
    Bt[idx] = (_Float16)w[((size_t)o * 256 + c) * 81 + p];
}
// dense weight [K][N] -> Bt [Npad][K] (zero pad rows n>=N)
__global__ void k_cvt_dense(const float* __restrict__ w, _Float16* __restrict__ Bt,
                            int K, int N, int Npad) {
    long idx = (long)blockIdx.x * 256 + threadIdx.x;
    if (idx >= (long)Npad * K) return;
    const int n = (int)(idx / K), k = (int)(idx % K);
    Bt[idx] = (_Float16)((n < N) ? w[(size_t)k * N + n] : 0.f);
}
// primary squash: preact[m=(b,s)][c=cap*8+a] -> x3[b][cap*36+s][a]
__global__ void k_squash_primary(const float* __restrict__ preact,
                                 const float* __restrict__ prim_b,
                                 float* __restrict__ x3) {
    int idx = blockIdx.x * 256 + threadIdx.x;
    if (idx >= BATCH * 36 * NPRIM) return;
    const int cap = idx % NPRIM;
    const int s   = (idx / NPRIM) % 36;
    const int b   = idx / (NPRIM * 36);
    const float* p = preact + ((size_t)(b * 36 + s)) * 256 + cap * 8;
    float v[8], n2 = 0.f;
    #pragma unroll
    for (int a = 0; a < 8; ++a) { v[a] = p[a] * (1.f / 32.f) + prim_b[cap * 8 + a]; n2 += v[a] * v[a]; }
    const float sc = (n2 / (1.f + n2)) * rsqrtf(n2 + 1e-9f);
    float* o = x3 + ((size_t)b * IN_DIM + cap * 36 + s) * 8;
    #pragma unroll
    for (int a = 0; a < 8; ++a) o[a] = v[a] * sc;
}
// mask digit caps by one-hot label -> f16 [256][160]
__global__ void k_mask(const float* __restrict__ digit, const float* __restrict__ y,
                       _Float16* __restrict__ masked) {
    int idx = blockIdx.x * 256 + threadIdx.x;
    if (idx >= BATCH * 160) return;
    const int b = idx / 160, j = idx % 160;
    masked[idx] = (_Float16)(digit[idx] * y[b * NCLS + j / 16]);
}

// ---------------------------------------------------------------------------
extern "C" void kernel_launch(void* const* d_in, const int* in_sizes, int n_in,
                              void* d_out, int out_size, void* d_ws, size_t ws_size,
                              hipStream_t stream) {
    const float* x       = (const float*)d_in[0];
    const float* y       = (const float*)d_in[1];
    const float* conv1_w = (const float*)d_in[2];
    const float* conv1_b = (const float*)d_in[3];
    const float* prim_w  = (const float*)d_in[4];
    const float* prim_b  = (const float*)d_in[5];
    const float* digit_w = (const float*)d_in[6];
    const float* digit_b = (const float*)d_in[7];
    const float* w1      = (const float*)d_in[8];
    const float* b1      = (const float*)d_in[9];
    const float* w2      = (const float*)d_in[10];
    const float* b2      = (const float*)d_in[11];
    const float* w3      = (const float*)d_in[12];
    const float* b3      = (const float*)d_in[13];

    float* out_digit = (float*)d_out;                 // [256][10][16]
    float* out_recon = (float*)d_out + BATCH * 160;   // [256][784]

    // workspace carve-out
    char* cur = (char*)d_ws;
    auto alloc = [&](size_t bytes) { void* p = cur; cur += (bytes + 255) & ~(size_t)255; return p; };
    _Float16* A1     = (_Float16*)alloc((size_t)M1 * K1 * 2);
    _Float16* B1t    = (_Float16*)alloc((size_t)256 * K1 * 2);
    _Float16* h      = (_Float16*)alloc((size_t)M1 * 256 * 2);      // NHWC conv1 out
    _Float16* Bpt    = (_Float16*)alloc((size_t)256 * K2 * 2);
    float*    preact = (float*)alloc((size_t)M2 * 256 * 4);
    float*    x3     = (float*)alloc((size_t)BATCH * IN_DIM * 8 * 4);
    float*    votes  = (float*)alloc((size_t)BATCH * IN_DIM * 160 * 4);
    float*    logits = (float*)alloc((size_t)BATCH * IN_DIM * NCLS * 4);
    _Float16* w1t    = (_Float16*)alloc((size_t)512 * 160 * 2);
    _Float16* w2t    = (_Float16*)alloc((size_t)1024 * 512 * 2);
    _Float16* w3t    = (_Float16*)alloc((size_t)832 * 1024 * 2);
    _Float16* masked = (_Float16*)alloc((size_t)BATCH * 160 * 2);
    _Float16* r1     = (_Float16*)alloc((size_t)BATCH * 512 * 2);
    _Float16* r2     = (_Float16*)alloc((size_t)BATCH * 1024 * 2);

    auto blks = [](long n) { return (unsigned)((n + 255) / 256); };

    // --- prep: weight converts, im2col, zero logits
    k_im2col1 <<<blks((long)M1 * K1), 256, 0, stream>>>(x, A1);
    k_cvt_w1  <<<blks(256 * K1), 256, 0, stream>>>(conv1_w, B1t);
    k_cvt_wp  <<<blks((long)256 * K2), 256, 0, stream>>>(prim_w, Bpt);
    k_cvt_dense<<<blks((long)512 * 160), 256, 0, stream>>>(w1, w1t, 160, 512, 512);
    k_cvt_dense<<<blks((long)1024 * 512), 256, 0, stream>>>(w2, w2t, 512, 1024, 1024);
    k_cvt_dense<<<blks((long)832 * 1024), 256, 0, stream>>>(w3, w3t, 1024, 784, 832);
    k_zero_f32<<<blks((long)BATCH * IN_DIM * NCLS), 256, 0, stream>>>(logits, BATCH * IN_DIM * NCLS);

    // --- conv1 as GEMM: [102400,96] x [96,256] -> bias+relu -> f16 NHWC h
    gemm_f16<0, 1><<<dim3(M1 / 128, 256 / 64), 256, 0, stream>>>(
        A1, B1t, conv1_b, nullptr, h, M1, 256, K1, 256, 256);

    // --- primary caps conv as implicit GEMM: [9216,20736] x [20736,256] -> raw f32
    gemm_f16<1, 0><<<dim3(M2 / 128, 256 / 64), 256, 0, stream>>>(
        h, Bpt, nullptr, preact, nullptr, M2, 256, K2, 256, 256);

    // --- primary squash -> x3 [B][1152][8]
    k_squash_primary<<<blks(BATCH * 36 * NPRIM), 256, 0, stream>>>(preact, prim_b, x3);

    // --- digit votes (fp32 WMMA 16x16x4): 1152*16*10 wave-jobs, 8 waves/block
    votes_wmma<<<(IN_DIM * 16 * NCLS) / 8, 256, 0, stream>>>(x3, digit_w, votes);

    // --- dynamic routing x3 (last iter skips logits update)
    routing_iter<<<BATCH, 512, 0, stream>>>(votes, digit_b, logits, out_digit, 1);
    routing_iter<<<BATCH, 512, 0, stream>>>(votes, digit_b, logits, out_digit, 1);
    routing_iter<<<BATCH, 512, 0, stream>>>(votes, digit_b, logits, out_digit, 0);

    // --- reconstruction MLP
    k_mask<<<blks(BATCH * 160), 256, 0, stream>>>(out_digit, y, masked);
    gemm_f16<0, 1><<<dim3(BATCH / 128, 512 / 64), 256, 0, stream>>>(
        masked, w1t, b1, nullptr, r1, BATCH, 512, 160, 512, 512);
    gemm_f16<0, 1><<<dim3(BATCH / 128, 1024 / 64), 256, 0, stream>>>(
        r1, w2t, b2, nullptr, r2, BATCH, 1024, 512, 1024, 1024);
    gemm_f16<0, 3><<<dim3(BATCH / 128, 832 / 64), 256, 0, stream>>>(
        r2, w3t, b3, out_recon, nullptr, BATCH, 832, 1024, 784, 784);
}
